// AdjustableLengthAttention_45354854646444
// MI455X (gfx1250) — matrix operand order repaired and checked by reading, not compiled, not run on hardware
//
#include <hip/hip_runtime.h>
#include <hip/hip_bf16.h>

typedef _Float16 v16h __attribute__((ext_vector_type(16)));
typedef _Float16 v8h  __attribute__((ext_vector_type(8)));
typedef float    v8f  __attribute__((ext_vector_type(8)));

#define B_  8
#define C_  512
#define CQ_ 64
#define H_  64
#define W_  64
#define N_  4096   // H_*W_

static __device__ __forceinline__ v8f wmma_f16(v16h a, v16h b, v8f c) {
  // D = A(16x32 f16) * B(32x16 f16) + C(16x16 f32)
  return __builtin_amdgcn_wmma_f32_16x16x32_f16(false, a, false, b, (short)0, c,
                                                false, false);
}

static __device__ __forceinline__ v8f vzero8() {
  v8f z;
#pragma unroll
  for (int r = 0; r < 8; ++r) z[r] = 0.0f;
  return z;
}

// A fragment (16x32, f16) from row-major matrix (K contiguous), tile origin at `base`.
// lanes 0-15: row m=lane, K = 0..7 and 16..23 ; lanes 16-31: row m=lane-16, K = 8..15 and 24..31
static __device__ __forceinline__ v16h load_a16(const _Float16* base, int lda, int lane) {
  const _Float16* p = base + (size_t)(lane & 15) * (size_t)lda + ((lane >> 4) << 3);
  v8h lo = *(const v8h*)p;
  v8h hi = *(const v8h*)(p + 16);
  v16h a;
#pragma unroll
  for (int i = 0; i < 8; ++i) { a[i] = lo[i]; a[i + 8] = hi[i]; }
  return a;
}

// B fragment (32x16, f16) from K-contiguous storage Bt[n][k] (row n = column of B).
// lanes 0-15: col n=lane, K = 0..7 and 16..23 ; lanes 16-31: col n=lane-16, K = 8..15 and 24..31
static __device__ __forceinline__ v16h load_b16(const _Float16* baseT, int ldb, int lane) {
  const _Float16* p = baseT + (size_t)(lane & 15) * (size_t)ldb + ((lane >> 4) << 3);
  v8h lo = *(const v8h*)p;
  v8h hi = *(const v8h*)(p + 16);
  v16h b;
#pragma unroll
  for (int i = 0; i < 8; ++i) { b[i] = lo[i]; b[i + 8] = hi[i]; }
  return b;
}

// Masked variants for the score GEMMs: zero element unless (row<len && k<len).
static __device__ __forceinline__ v16h load_a16_mask(const _Float16* mat, int lda, int lane,
                                                     int m0, int k0, int len) {
  const int m  = m0 + (lane & 15);
  const int kb = k0 + ((lane >> 4) << 3);
  const _Float16* p = mat + (size_t)m * (size_t)lda + kb;
  v8h lo = *(const v8h*)p;
  v8h hi = *(const v8h*)(p + 16);
  const bool rowok = m < len;
  v16h a;
#pragma unroll
  for (int i = 0; i < 8; ++i) {
    a[i]     = (rowok && (kb + i)      < len) ? lo[i] : (_Float16)0.0f;
    a[i + 8] = (rowok && (kb + 16 + i) < len) ? hi[i] : (_Float16)0.0f;
  }
  return a;
}

static __device__ __forceinline__ v16h load_b16_mask(const _Float16* mat, int ldb, int lane,
                                                     int n0, int k0, int len) {
  const int n  = n0 + (lane & 15);
  const int kb = k0 + ((lane >> 4) << 3);
  const _Float16* p = mat + (size_t)n * (size_t)ldb + kb;
  v8h lo = *(const v8h*)p;
  v8h hi = *(const v8h*)(p + 16);
  const bool colok = n < len;
  v16h b;
#pragma unroll
  for (int i = 0; i < 8; ++i) {
    b[i]     = (colok && (kb + i)     < len) ? lo[i] : (_Float16)0.0f;
    b[i + 8] = (colok && (kb + 16 + i) < len) ? hi[i] : (_Float16)0.0f;
  }
  return b;
}

// ---------------------------------------------------------------------------
// Weights fp32 -> f16
__global__ void __launch_bounds__(256) cvt_w_kernel(
    const float* __restrict__ Wq, const float* __restrict__ Wk,
    const float* __restrict__ Wv, _Float16* __restrict__ wq,
    _Float16* __restrict__ wk, _Float16* __restrict__ wv) {
  const int i = blockIdx.x * 256 + threadIdx.x;
  if (i < CQ_ * C_) { wq[i] = (_Float16)Wq[i]; wk[i] = (_Float16)Wk[i]; }
  if (i < C_ * C_)  { wv[i] = (_Float16)Wv[i]; }
}

// ---------------------------------------------------------------------------
// Projection: Y[o,n] = sum_c Wf[o,c] * X[b,c,n] + bias[o]
// grid = (N_/128, M/64, B_), block = 256 (8 waves). Per block: 64(M) x 128(N).
// Outputs:
//   outNC  (q,k): [b][n][M]   (M contiguous)      -- score GEMM fragments
//   outHW  (v)  : [b][c][h][w] (w contiguous)     -- out2 A fragments
//   outWH  (v)  : [b][c][w][h] (h contiguous)     -- out1 A fragments
__global__ void __launch_bounds__(256) proj_kernel(
    const float* __restrict__ X, const _Float16* __restrict__ Wf,
    const float* __restrict__ bias, int M,
    _Float16* __restrict__ outNC, _Float16* __restrict__ outHW,
    _Float16* __restrict__ outWH) {
  __shared__ _Float16 ldsB[128 * 40];   // Bt[n_local][k_local], pitch 40 halves (80B)
  const int b  = blockIdx.z;
  const int n0 = blockIdx.x * 128;
  const int m0 = blockIdx.y * 64;
  const int t = threadIdx.x, lane = t & 31, wv = t >> 5;
  const int wm = (wv & 3) * 16;   // wave M offset within block tile
  const int wn = (wv >> 2) * 64;  // wave N offset within block tile

  v8f acc[4];
#pragma unroll
  for (int nt = 0; nt < 4; ++nt) acc[nt] = vzero8();

  const int nl = t & 127, kh = t >> 7;  // staging role: column nl, k-half kh
  for (int kc = 0; kc < C_; kc += 32) {
    // stage X[kc..kc+31][n0..n0+127] fp32 -> f16, K-contiguous in LDS
    const float* xp = X + ((size_t)b * C_ + kc + kh * 16) * N_ + n0 + nl;
    _Float16* lp = ldsB + nl * 40 + kh * 16;
#pragma unroll
    for (int j = 0; j < 16; ++j) lp[j] = (_Float16)xp[(size_t)j * N_];
    __syncthreads();

    const v16h af = load_a16(Wf + (size_t)(m0 + wm) * C_ + kc, C_, lane);
#pragma unroll
    for (int nt = 0; nt < 4; ++nt) {
      v16h bf = load_b16(ldsB + (wn + nt * 16) * 40, 40, lane);
      acc[nt] = wmma_f16(af, bf, acc[nt]);
    }
    __syncthreads();
  }

  const int mrow = m0 + wm + ((lane >> 4) << 3);  // first of this lane's 8 rows
  float bvv[8];
#pragma unroll
  for (int r = 0; r < 8; ++r) bvv[r] = bias[mrow + r];

#pragma unroll
  for (int nt = 0; nt < 4; ++nt) {
    const int n = n0 + wn + nt * 16 + (lane & 15);
    if (outNC) {
      v8h h8;
#pragma unroll
      for (int r = 0; r < 8; ++r) h8[r] = (_Float16)(acc[nt][r] + bvv[r]);
      *(v8h*)(outNC + ((size_t)b * N_ + n) * (size_t)M + mrow) = h8;
    }
    if (outHW) {
      const int nwh = (n & 63) * 64 + (n >> 6);  // (w,h) spatial index
#pragma unroll
      for (int r = 0; r < 8; ++r) {
        const _Float16 val = (_Float16)(acc[nt][r] + bvv[r]);
        outHW[(size_t)(b * M + mrow + r) * N_ + n]   = val;
        outWH[(size_t)(b * M + mrow + r) * N_ + nwh] = val;
      }
    }
  }
}

// ---------------------------------------------------------------------------
// Scores: one wave per instance.
//  inst <  512 : H-branch (b,w): SH[b][w][h][g] = sum_{c<len} q[c,h,w]k[c,g,w] masked
//  inst >= 512 : W-branch (b,h): SW[b][h][w][g] = sum_{c<len} q[c,h,w]k[c,h,g] masked
__global__ void __launch_bounds__(256) score_kernel(
    const _Float16* __restrict__ q, const _Float16* __restrict__ k,
    float* __restrict__ SH, float* __restrict__ SW, const int* __restrict__ lenp) {
  const int lane = threadIdx.x & 31;
  const int inst = blockIdx.x * 8 + (threadIdx.x >> 5);
  const int len = *lenp;
  const int isW = inst >> 9;
  const int ii = inst & 511;
  const int b = ii >> 6, s = ii & 63;
  size_t off; int ld; float* Sout;
  if (!isW) {  // rows h, stride between h = W_*CQ_ halves; fixed w = s
    off = (size_t)b * ((size_t)N_ * CQ_) + (size_t)s * CQ_;
    ld = W_ * CQ_;
    Sout = SH + (size_t)ii * 4096;
  } else {     // rows w, stride CQ_ halves; fixed h = s
    off = (size_t)b * ((size_t)N_ * CQ_) + (size_t)s * (W_ * CQ_);
    ld = CQ_;
    Sout = SW + (size_t)ii * 4096;
  }
  const _Float16* qb = q + off;
  const _Float16* kb = k + off;
#pragma unroll
  for (int mt = 0; mt < 4; ++mt) {
#pragma unroll
    for (int nt = 0; nt < 4; ++nt) {
      v8f acc = vzero8();
#pragma unroll
      for (int ks = 0; ks < CQ_; ks += 32) {
        v16h af = load_a16_mask(qb, ld, lane, mt * 16, ks, len);
        v16h bf = load_b16_mask(kb, ld, lane, nt * 16, ks, len);
        acc = wmma_f16(af, bf, acc);
      }
      const int col = nt * 16 + (lane & 15);
      const int row = mt * 16 + ((lane >> 4) << 3);
#pragma unroll
      for (int r = 0; r < 8; ++r) Sout[(size_t)(row + r) * 64 + col] = acc[r];
    }
  }
}

// ---------------------------------------------------------------------------
// Softmax over concat([eH (diag -inf), eW]) per (b,h,w); writes f16 attention
// in Bt layouts: AH[b][w][h][g], AW[b][h][w][g].
__global__ void __launch_bounds__(256) softmax_kernel(
    const float* __restrict__ SH, const float* __restrict__ SW,
    _Float16* __restrict__ AH, _Float16* __restrict__ AW) {
  const int tid = blockIdx.x * 256 + threadIdx.x;  // 0..32767
  const int b = tid >> 12, h = (tid >> 6) & 63, w = tid & 63;
  const float* sh = SH + (size_t)((b * 64 + w) * 64 + h) * 64;
  const float* sw = SW + (size_t)((b * 64 + h) * 64 + w) * 64;
  float lh[64], lw[64];
#pragma unroll
  for (int g = 0; g < 64; ++g) lh[g] = (g == h) ? -__builtin_inff() : sh[g];
#pragma unroll
  for (int g = 0; g < 64; ++g) lw[g] = sw[g];
  float mx = lw[0];
#pragma unroll
  for (int g = 0; g < 64; ++g) { mx = fmaxf(mx, lh[g]); mx = fmaxf(mx, lw[g]); }
  float s = 0.0f;
#pragma unroll
  for (int g = 0; g < 64; ++g) { lh[g] = __expf(lh[g] - mx); s += lh[g]; }
#pragma unroll
  for (int g = 0; g < 64; ++g) { lw[g] = __expf(lw[g] - mx); s += lw[g]; }
  const float inv = 1.0f / s;
  _Float16* ah = AH + (size_t)((b * 64 + w) * 64 + h) * 64;
  _Float16* aw = AW + (size_t)((b * 64 + h) * 64 + w) * 64;
#pragma unroll
  for (int g = 0; g < 64; ++g) ah[g] = (_Float16)(lh[g] * inv);
#pragma unroll
  for (int g = 0; g < 64; ++g) aw[g] = (_Float16)(lw[g] * inv);
}

// ---------------------------------------------------------------------------
// out1 per (b,w): O1[c,i] = sum_j v[b,c,j,w] * AH[b,w,i,j]   (A = v_wh, B = AH)
// grid = (512, 4); block 256 (8 waves); wave tile = 16(c) x 64(i).
__global__ void __launch_bounds__(256) out1_kernel(
    const _Float16* __restrict__ vWH, const _Float16* __restrict__ AH,
    float* __restrict__ O1) {
  const int bw = blockIdx.x;                 // b*64 + w
  const int b = bw >> 6, w = bw & 63;
  const int lane = threadIdx.x & 31, wv = threadIdx.x >> 5;
  const int c0 = blockIdx.y * 128 + wv * 16;
  const _Float16* ab = vWH + (size_t)(b * C_ + c0) * N_ + w * 64;  // A(c,j), lda=N_
  const _Float16* bb = AH + (size_t)bw * 4096;                     // Bt(i,j), ldb=64
  v8f acc[4];
#pragma unroll
  for (int nt = 0; nt < 4; ++nt) acc[nt] = vzero8();
#pragma unroll
  for (int ks = 0; ks < 64; ks += 32) {
    const v16h af = load_a16(ab + ks, N_, lane);
#pragma unroll
    for (int nt = 0; nt < 4; ++nt) {
      v16h bf = load_b16(bb + nt * 16 * 64 + ks, 64, lane);
      acc[nt] = wmma_f16(af, bf, acc[nt]);
    }
  }
  const int crow = c0 + ((lane >> 4) << 3);
#pragma unroll
  for (int nt = 0; nt < 4; ++nt) {
    const int i = nt * 16 + (lane & 15);
#pragma unroll
    for (int r = 0; r < 8; ++r)
      O1[(size_t)(b * C_ + crow + r) * N_ + i * 64 + w] = acc[nt][r];
  }
}

// out2 per (b,h): O2[c,i] = sum_j v[b,c,h,j] * AW[b,h,i,j]; then
// out = gamma*(O1+O2) + x1.
__global__ void __launch_bounds__(256) out2_kernel(
    const _Float16* __restrict__ vHW, const _Float16* __restrict__ AW,
    const float* __restrict__ O1, const float* __restrict__ x1,
    const float* __restrict__ gammap, float* __restrict__ out) {
  const int bh = blockIdx.x;                 // b*64 + h
  const int b = bh >> 6, h = bh & 63;
  const int lane = threadIdx.x & 31, wv = threadIdx.x >> 5;
  const int c0 = blockIdx.y * 128 + wv * 16;
  const _Float16* ab = vHW + (size_t)(b * C_ + c0) * N_ + h * 64;  // A(c,j), lda=N_
  const _Float16* bb = AW + (size_t)bh * 4096;                     // Bt(i,j), ldb=64
  v8f acc[4];
#pragma unroll
  for (int nt = 0; nt < 4; ++nt) acc[nt] = vzero8();
#pragma unroll
  for (int ks = 0; ks < 64; ks += 32) {
    const v16h af = load_a16(ab + ks, N_, lane);
#pragma unroll
    for (int nt = 0; nt < 4; ++nt) {
      v16h bf = load_b16(bb + nt * 16 * 64 + ks, 64, lane);
      acc[nt] = wmma_f16(af, bf, acc[nt]);
    }
  }
  const float gma = *gammap;
  const int crow = c0 + ((lane >> 4) << 3);
#pragma unroll
  for (int nt = 0; nt < 4; ++nt) {
    const int i = nt * 16 + (lane & 15);
#pragma unroll
    for (int r = 0; r < 8; ++r) {
      const size_t idx = (size_t)(b * C_ + crow + r) * N_ + h * 64 + i;
      out[idx] = gma * (O1[idx] + acc[nt][r]) + x1[idx];
    }
  }
}

// ---------------------------------------------------------------------------
extern "C" void kernel_launch(void* const* d_in, const int* in_sizes, int n_in,
                              void* d_out, int out_size, void* d_ws, size_t ws_size,
                              hipStream_t stream) {
  (void)in_sizes; (void)n_in; (void)out_size; (void)ws_size;
  const float* x1 = (const float*)d_in[0];
  const float* x2 = (const float*)d_in[1];
  const float* x3 = (const float*)d_in[2];
  const float* Wq = (const float*)d_in[3];
  const float* bq = (const float*)d_in[4];
  const float* Wk = (const float*)d_in[5];
  const float* bk = (const float*)d_in[6];
  const float* Wv = (const float*)d_in[7];
  const float* bv = (const float*)d_in[8];
  const float* gamma = (const float*)d_in[9];
  const int*   len = (const int*)d_in[10];
  float* out = (float*)d_out;

  char* wsp = (char*)d_ws;
  size_t off = 0;
  auto carve = [&](size_t bytes) -> void* {
    void* p = wsp + off;
    off += (bytes + 255) & ~(size_t)255;
    return p;
  };
  _Float16* qnc  = (_Float16*)carve((size_t)B_ * N_ * CQ_ * 2);   // [b][n][cq]
  _Float16* knc  = (_Float16*)carve((size_t)B_ * N_ * CQ_ * 2);   // [b][n][cq]
  _Float16* vhw  = (_Float16*)carve((size_t)B_ * C_ * N_ * 2);    // [b][c][h][w]
  _Float16* vwh  = (_Float16*)carve((size_t)B_ * C_ * N_ * 2);    // [b][c][w][h]
  float*    SH   = (float*)   carve((size_t)B_ * 64 * 64 * 64 * 4);
  float*    SW   = (float*)   carve((size_t)B_ * 64 * 64 * 64 * 4);
  _Float16* AH   = (_Float16*)carve((size_t)B_ * 64 * 64 * 64 * 2);
  _Float16* AW   = (_Float16*)carve((size_t)B_ * 64 * 64 * 64 * 2);
  float*    O1   = (float*)   carve((size_t)B_ * C_ * N_ * 4);
  _Float16* wq16 = (_Float16*)carve((size_t)CQ_ * C_ * 2);
  _Float16* wk16 = (_Float16*)carve((size_t)CQ_ * C_ * 2);
  _Float16* wv16 = (_Float16*)carve((size_t)C_ * C_ * 2);

  cvt_w_kernel<<<1024, 256, 0, stream>>>(Wq, Wk, Wv, wq16, wk16, wv16);
  proj_kernel<<<dim3(32, 1, 8), 256, 0, stream>>>(x1, wq16, bq, CQ_, qnc, nullptr, nullptr);
  proj_kernel<<<dim3(32, 1, 8), 256, 0, stream>>>(x2, wk16, bk, CQ_, knc, nullptr, nullptr);
  proj_kernel<<<dim3(32, 8, 8), 256, 0, stream>>>(x3, wv16, bv, C_, nullptr, vhw, vwh);
  score_kernel<<<128, 256, 0, stream>>>(qnc, knc, SH, SW, len);
  softmax_kernel<<<128, 256, 0, stream>>>(SH, SW, AH, AW);
  out1_kernel<<<dim3(512, 4), 256, 0, stream>>>(vwh, AH, O1);
  out2_kernel<<<dim3(512, 4), 256, 0, stream>>>(vhw, AW, O1, x1, gamma, out);
}
